// OnlineFlashMTPModel_27152783245701
// MI455X (gfx1250) — matrix operand order repaired and checked by reading, not compiled, not run on hardware
//
#include <hip/hip_runtime.h>
#include <cstdint>
#include <cstddef>

// ---- problem constants (match reference) ----
#define BB      2
#define SS      4096
#define HDIM    2048
#define VOCAB   32000
#define NHH     16
#define DHH     128
#define BLK     16          // BS_BLK
#define NA      128         // N_ANCH
#define LL      2           // context layers
#define MASKID  31999

typedef float  v8f   __attribute__((ext_vector_type(8)));
typedef __bf16 v16bf __attribute__((ext_vector_type(16)));

union FragBF { v16bf v; uint4 q[2]; };

// ---- CDNA5 async global->LDS path ----
typedef int v4i __attribute__((ext_vector_type(4)));
typedef v4i __attribute__((address_space(1))) as1_v4i;   // global int4
typedef v4i __attribute__((address_space(3))) as3_v4i;   // LDS int4
typedef __attribute__((address_space(3))) void as3_void_t;

#if defined(__has_builtin)
#  if __has_builtin(__builtin_amdgcn_global_load_async_to_lds_b128)
#    define ASYNC_TIER 1
#  endif
#  if __has_builtin(__builtin_amdgcn_s_wait_asynccnt)
#    define HAVE_WAIT_ASYNC_BUILTIN 1
#  endif
#endif
#ifndef ASYNC_TIER
#  define ASYNC_TIER 2      // inline-asm tier; set to 3 to force sync fallback
#endif

__device__ __forceinline__ void async_ld16(const __bf16* g, __bf16* l)
{
#if ASYNC_TIER == 1
    __builtin_amdgcn_global_load_async_to_lds_b128(
        (as1_v4i*)g, (as3_v4i*)l, 0, 0);
#elif ASYNC_TIER == 2
    unsigned loff = (unsigned)(uintptr_t)(as3_void_t*)l;   // LDS byte address
    asm volatile("global_load_async_to_lds_b128 %0, %1, off"
                 :: "v"(loff), "v"(g) : "memory");
#endif
}

__device__ __forceinline__ void wait_async0()
{
#if defined(HAVE_WAIT_ASYNC_BUILTIN)
    __builtin_amdgcn_s_wait_asynccnt(0);
#else
    asm volatile("s_wait_asynccnt 0x0" ::: "memory");
#endif
}

// =====================================================================
// 1) Transpose f32 (K x N) weight -> bf16 (N x K)  (B^T for WMMA B-frags)
// =====================================================================
__global__ __launch_bounds__(256) void transpose_to_bf16_kernel(
    const float* __restrict__ W, __bf16* __restrict__ WT, int K, int N)
{
    __shared__ float tile[32][33];
    const int kb = blockIdx.x * 32, nb = blockIdx.y * 32;
    const int tx = threadIdx.x & 31, ty = threadIdx.x >> 5;   // 32 x 8
#pragma unroll
    for (int r = 0; r < 4; ++r) {
        int k = kb + ty + r * 8;
        tile[ty + r * 8][tx] = W[(size_t)k * N + nb + tx];
    }
    __syncthreads();
#pragma unroll
    for (int r = 0; r < 4; ++r) {
        int n = nb + ty + r * 8;
        WT[(size_t)n * K + kb + tx] = (__bf16)tile[tx][ty + r * 8];
    }
}

// =====================================================================
// 2) Anchor selection: stable top-128-smallest rand_vals, sorted ascending
// =====================================================================
__global__ __launch_bounds__(256) void anchor_kernel(
    const int*   __restrict__ input_ids,
    const float* __restrict__ loss_mask,
    const float* __restrict__ rand_vals,
    int* __restrict__ anch, int* __restrict__ keep,
    int* __restrict__ tokm, int* __restrict__ ctx)
{
    __shared__ float svals[4096];
    __shared__ float rv[256];
    __shared__ int   ri[256];
    __shared__ int   spick[NA];
    const int b = blockIdx.x, tid = threadIdx.x;
    const int NVAL = SS - BLK + 1;   // 4081

    for (int i = tid; i < 4096; i += 256) {
        float v = 4.0f;
        if (i < NVAL) {
            bool valid = loss_mask[b * SS + i] > 0.5f;
            v = valid ? rand_vals[b * NVAL + i] : 2.0f;   // invalid -> 2.0 (ref)
        }
        svals[i] = v;
    }
    __syncthreads();

    for (int round = 0; round < NA; ++round) {
        float best = 5.0f; int bi = 0x7fffffff;
        for (int i = tid; i < 4096; i += 256) {
            float v = svals[i];
            if (v < best) { best = v; bi = i; }            // first-hit = smallest idx
        }
        rv[tid] = best; ri[tid] = bi;
        __syncthreads();
        for (int s = 128; s > 0; s >>= 1) {
            if (tid < s) {
                float v2 = rv[tid + s]; int i2 = ri[tid + s];
                if (v2 < rv[tid] || (v2 == rv[tid] && i2 < ri[tid])) { rv[tid] = v2; ri[tid] = i2; }
            }
            __syncthreads();
        }
        if (tid == 0) {
            int w = ri[0];
            spick[round] = (svals[w] < 1.5f) ? w : (SS + 1);  // masked_idx semantics
            svals[w] = 4.0f;
        }
        __syncthreads();
    }

    // bitonic ascending sort of the 128 picks
    for (int k = 2; k <= NA; k <<= 1) {
        for (int j = k >> 1; j > 0; j >>= 1) {
            if (tid < NA) {
                int ixj = tid ^ j;
                if (ixj > tid) {
                    int a = spick[tid], c = spick[ixj];
                    bool up = ((tid & k) == 0);
                    if ((a > c) == up) { spick[tid] = c; spick[ixj] = a; }
                }
            }
            __syncthreads();
        }
    }

    if (tid < NA) {
        int a  = spick[tid];
        int kp = (a != SS + 1) ? 1 : 0;
        int an = kp ? a : 0;
        int tk = input_ids[b * SS + an];
        int o  = b * NA + tid;
        anch[o] = an;
        keep[o] = kp;
        tokm[o] = kp ? tk : MASKID;
        ctx[o]  = (an > 0) ? (an - 1) : 0;
    }
}

// =====================================================================
// 3) Gather embedding / hidden rows into bf16 A-matrices
// =====================================================================
__global__ __launch_bounds__(256) void build_rows_kernel(
    const float* __restrict__ embed, const float* __restrict__ hidden,
    const int* __restrict__ tokm, const int* __restrict__ ctx,
    __bf16* __restrict__ noiseA, __bf16* __restrict__ kvA)
{
    const int row = blockIdx.x, tid = threadIdx.x;
    const int NQ = BB * NA * BLK;
    const float* src;
    __bf16* dst;
    if (row < NQ) {
        int bn = row / BLK, t = row % BLK;
        int tok = (t == 0) ? tokm[bn] : MASKID;
        src = embed + (size_t)tok * HDIM;
        dst = noiseA + (size_t)row * HDIM;
    } else {
        int r2 = row - NQ;
        int bn = r2 / (LL + BLK), s = r2 % (LL + BLK);
        if (s < LL) {
            int lid = (s == 0) ? 1 : 2;            // LAYER_IDS = (1,2)
            int b   = bn / NA;
            src = hidden + (((size_t)lid * BB + b) * SS + ctx[bn]) * HDIM;
        } else {
            int t   = s - LL;
            int tok = (t == 0) ? tokm[bn] : MASKID;
            src = embed + (size_t)tok * HDIM;
        }
        dst = kvA + (size_t)r2 * HDIM;
    }
    for (int i = tid; i < HDIM; i += 256) dst[i] = (__bf16)src[i];
}

// =====================================================================
// 4) WMMA bf16 GEMM: C(MxN,f32) = A(MxK,bf16) * BT(NxK,bf16)^T
//    64x64 block tile, BK=32, 4 waves (2x2) of 32x32.
//    Double-buffered LDS fed by GLOBAL_LOAD_ASYNC_TO_LDS_B128 (ASYNCcnt),
//    so tile t+1 streams into LDS while tile t runs on the WMMA pipe.
//    EPI==0: store f32 C.  EPI==1: +embed residual, *keep, store bf16.
// =====================================================================
template <int EPI>
__global__ __launch_bounds__(128) void gemm_bf16_kernel(
    const __bf16* __restrict__ A, const __bf16* __restrict__ BT,
    float* __restrict__ C, __bf16* __restrict__ Cbf,
    int M, int N, int K,
    const int* __restrict__ tokm, const int* __restrict__ keep,
    const float* __restrict__ embed)
{
    constexpr int LDT = 40;                 // padded LDS row stride (bf16)
    __shared__ __bf16 As[2][64 * LDT];
    __shared__ __bf16 Bs[2][64 * LDT];

    const int tid  = threadIdx.x;
    const int lane = tid & 31, wave = tid >> 5;
    const int half = lane >> 4, l15 = lane & 15;
    const int wm = wave >> 1, wn = wave & 1;
    const int bm = blockIdx.y * 64, bn = blockIdx.x * 64;

    // per-thread staging slots: two 16B chunks per matrix per tile
    const int row0 = tid >> 2, off0 = (tid & 3) * 8;   // rows 0..31
    const int row1 = row0 + 32;                        // rows 32..63
    const __bf16* Ag0 = A  + (size_t)(bm + row0) * K + off0;
    const __bf16* Ag1 = A  + (size_t)(bm + row1) * K + off0;
    const __bf16* Bg0 = BT + (size_t)(bn + row0) * K + off0;
    const __bf16* Bg1 = BT + (size_t)(bn + row1) * K + off0;
    const int la0 = row0 * LDT + off0;
    const int la1 = row1 * LDT + off0;

    v8f c[2][2] = {};
    int p = 0;

#if ASYNC_TIER <= 2
    // ---------------- async double-buffered pipeline ----------------
    async_ld16(Ag0, &As[0][la0]);
    async_ld16(Ag1, &As[0][la1]);
    async_ld16(Bg0, &Bs[0][la0]);
    async_ld16(Bg1, &Bs[0][la1]);
    for (int kt = 0; kt < K; kt += 32) {
        wait_async0();              // this wave's tile-kt loads landed in LDS
        __syncthreads();            // every wave's loads landed; prev reads done
        if (kt + 32 < K) {          // stream tile kt+32 while computing kt
            async_ld16(Ag0 + kt + 32, &As[p ^ 1][la0]);
            async_ld16(Ag1 + kt + 32, &As[p ^ 1][la1]);
            async_ld16(Bg0 + kt + 32, &Bs[p ^ 1][la0]);
            async_ld16(Bg1 + kt + 32, &Bs[p ^ 1][la1]);
        }
#else
    // ---------------- synchronous register-staged fallback ----------------
    uint4 ra0 = *(const uint4*)Ag0, ra1 = *(const uint4*)Ag1;
    uint4 rb0 = *(const uint4*)Bg0, rb1 = *(const uint4*)Bg1;
    for (int kt = 0; kt < K; kt += 32) {
        __syncthreads();            // buffer p free (compute 2 iters ago done)
        *(uint4*)(&As[p][la0]) = ra0;
        *(uint4*)(&As[p][la1]) = ra1;
        *(uint4*)(&Bs[p][la0]) = rb0;
        *(uint4*)(&Bs[p][la1]) = rb1;
        __syncthreads();
        if (kt + 32 < K) {          // issue next tile's loads before compute
            ra0 = *(const uint4*)(Ag0 + kt + 32);
            ra1 = *(const uint4*)(Ag1 + kt + 32);
            rb0 = *(const uint4*)(Bg0 + kt + 32);
            rb1 = *(const uint4*)(Bg1 + kt + 32);
        }
#endif
        FragBF a[2], bfr[2];
#pragma unroll
        for (int i = 0; i < 2; ++i) {
            // A 16x32 bf16: lanes0-15 K0-7 & K16-23, lanes16-31 K8-15 & K24-31
            int ar = wm * 32 + i * 16 + l15;
            a[i].q[0] = *(const uint4*)(&As[p][ar * LDT + 8 * half]);
            a[i].q[1] = *(const uint4*)(&As[p][ar * LDT + 16 + 8 * half]);
            // B 32x16 bf16: lane = col n; lanes0-15 K0-7 & K16-23, lanes16-31 K8-15 & K24-31
            int br = wn * 32 + i * 16 + l15;
            bfr[i].q[0] = *(const uint4*)(&Bs[p][br * LDT + 8 * half]);
            bfr[i].q[1] = *(const uint4*)(&Bs[p][br * LDT + 16 + 8 * half]);
        }
#pragma unroll
        for (int i = 0; i < 2; ++i)
#pragma unroll
            for (int j = 0; j < 2; ++j)
                c[i][j] = __builtin_amdgcn_wmma_f32_16x16x32_bf16(
                    false, a[i].v, false, bfr[j].v, (short)0, c[i][j], false, false);
        p ^= 1;
    }

    // epilogue: C/D layout = VGPR r, lane L -> M = r + 8*(L>>4), N = L&15
#pragma unroll
    for (int i = 0; i < 2; ++i) {
#pragma unroll
        for (int j = 0; j < 2; ++j) {
            int n0 = bn + wn * 32 + j * 16 + l15;
#pragma unroll
            for (int r = 0; r < 8; ++r) {
                int mm  = bm + wm * 32 + i * 16 + r + 8 * half;
                float v = c[i][j][r];
                if (EPI == 0) {
                    C[(size_t)mm * N + n0] = v;
                } else {
                    int bnn = mm >> 4, t = mm & 15;
                    int tok = (t == 0) ? tokm[bnn] : MASKID;
                    float kpv = keep[bnn] ? 1.0f : 0.0f;
                    v = (v + embed[(size_t)tok * HDIM + n0]) * kpv;
                    Cbf[(size_t)mm * N + n0] = (__bf16)v;
                }
            }
        }
    }
}

// =====================================================================
// 5) RoPE in place on f32 q/k: pairs (d, d+64) within each DH=128 head
// =====================================================================
__global__ __launch_bounds__(256) void rope_kernel(
    float* __restrict__ x, int T, int is_k,
    const int* __restrict__ anch, const int* __restrict__ ctx)
{
    int gid = blockIdx.x * 256 + threadIdx.x;
    int d  = gid & 63;
    int h  = (gid >> 6) & (NHH - 1);
    int rt = gid >> 10;
    int t  = rt % T;
    int bn = rt / T;
    if (bn >= BB * NA) return;
    int pos = !is_k ? (anch[bn] + t)
                    : ((t < LL) ? ctx[bn] : (anch[bn] + (t - LL)));
    float inv = __powf(10000.0f, -(float)d * (1.0f / 64.0f));
    float ang = (float)pos * inv;
    float s, cth;
    __sincosf(ang, &s, &cth);
    size_t base = ((size_t)(bn * T + t)) * HDIM + h * DHH + d;
    float x1 = x[base], x2 = x[base + 64];
    x[base]      = x1 * cth - x2 * s;
    x[base + 64] = x2 * cth + x1 * s;
}

// =====================================================================
// 6) Attention per (b, anchor, head): 16x18 softmax(QK^T/sqrt(128)) * V
// =====================================================================
__global__ __launch_bounds__(128) void attn_kernel(
    const float* __restrict__ qf, const float* __restrict__ kf,
    const float* __restrict__ vf, const int* __restrict__ keep,
    __bf16* __restrict__ oA)
{
    __shared__ float qs[BLK * DHH];
    __shared__ float ks[(LL + BLK) * DHH];
    __shared__ float vs[(LL + BLK) * DHH];
    __shared__ float sc[BLK * (LL + BLK)];
    const int id = blockIdx.x, tid = threadIdx.x;
    const int h = id & (NHH - 1), bn = id >> 4;

    const float* qb = qf + (size_t)(bn * BLK) * HDIM + h * DHH;
    const float* kb = kf + (size_t)(bn * (LL + BLK)) * HDIM + h * DHH;
    const float* vb = vf + (size_t)(bn * (LL + BLK)) * HDIM + h * DHH;
    for (int e = tid; e < BLK * DHH; e += 128)
        qs[e] = qb[(size_t)(e >> 7) * HDIM + (e & 127)];
    for (int e = tid; e < (LL + BLK) * DHH; e += 128) {
        ks[e] = kb[(size_t)(e >> 7) * HDIM + (e & 127)];
        vs[e] = vb[(size_t)(e >> 7) * HDIM + (e & 127)];
    }
    __syncthreads();

    const int kp = keep[bn];
    for (int s = tid; s < BLK * (LL + BLK); s += 128) {
        int i = s / (LL + BLK), j = s % (LL + BLK);
        float acc = 0.f;
        for (int d = 0; d < DHH; ++d) acc += qs[i * DHH + d] * ks[j * DHH + d];
        sc[s] = kp ? acc * 0.08838834764831845f : -1.0e9f;   // 1/sqrt(128)
    }
    __syncthreads();

    if (tid < BLK) {
        float m = -3.0e38f;
        for (int j = 0; j < LL + BLK; ++j) m = fmaxf(m, sc[tid * (LL + BLK) + j]);
        float ssum = 0.f;
        for (int j = 0; j < LL + BLK; ++j) {
            float e = __expf(sc[tid * (LL + BLK) + j] - m);
            sc[tid * (LL + BLK) + j] = e; ssum += e;
        }
        float r = 1.0f / ssum;
        for (int j = 0; j < LL + BLK; ++j) sc[tid * (LL + BLK) + j] *= r;
    }
    __syncthreads();

    const int d = tid;   // 128 threads = 128 head dims
    for (int i = 0; i < BLK; ++i) {
        float acc = 0.f;
        for (int j = 0; j < LL + BLK; ++j) acc += sc[i * (LL + BLK) + j] * vs[j * DHH + d];
        oA[(size_t)(bn * BLK + i) * HDIM + h * DHH + d] = (__bf16)acc;
    }
}

// =====================================================================
// host-side launcher
// =====================================================================
extern "C" void kernel_launch(void* const* d_in, const int* in_sizes, int n_in,
                              void* d_out, int out_size, void* d_ws, size_t ws_size,
                              hipStream_t stream)
{
    (void)in_sizes; (void)n_in; (void)out_size; (void)ws_size;
    const int*   input_ids = (const int*)  d_in[0];
    const float* hidden    = (const float*)d_in[1];
    const float* loss_mask = (const float*)d_in[2];
    const float* rand_vals = (const float*)d_in[3];
    const float* embed     = (const float*)d_in[4];
    const float* Wq        = (const float*)d_in[5];
    const float* Wk        = (const float*)d_in[6];
    const float* Wv        = (const float*)d_in[7];
    const float* Wo        = (const float*)d_in[8];
    const float* Wlm       = (const float*)d_in[9];
    float* out = (float*)d_out;

    char* ws = (char*)d_ws;
    size_t off = 0;
    auto alloc = [&](size_t bytes) -> char* {
        char* p = ws + off;
        off = (off + bytes + 255) & ~(size_t)255;
        return p;
    };

    const int MQ  = BB * NA * BLK;          // 4096
    const int MKV = BB * NA * (LL + BLK);   // 4608

    int* anch = (int*)alloc((size_t)BB * NA * 4);
    int* keep = (int*)alloc((size_t)BB * NA * 4);
    int* tokm = (int*)alloc((size_t)BB * NA * 4);
    int* ctx  = (int*)alloc((size_t)BB * NA * 4);
    __bf16* WqT  = (__bf16*)alloc((size_t)HDIM * HDIM * 2);
    __bf16* WkT  = (__bf16*)alloc((size_t)HDIM * HDIM * 2);
    __bf16* WvT  = (__bf16*)alloc((size_t)HDIM * HDIM * 2);
    __bf16* WoT  = (__bf16*)alloc((size_t)HDIM * HDIM * 2);
    __bf16* WlmT = (__bf16*)alloc((size_t)VOCAB * HDIM * 2);
    __bf16* noiseA = (__bf16*)alloc((size_t)MQ  * HDIM * 2);
    __bf16* kvA    = (__bf16*)alloc((size_t)MKV * HDIM * 2);
    float* qf = (float*)alloc((size_t)MQ  * HDIM * 4);
    float* kf = (float*)alloc((size_t)MKV * HDIM * 4);
    float* vf = (float*)alloc((size_t)MKV * HDIM * 4);
    __bf16* oA   = (__bf16*)alloc((size_t)MQ * HDIM * 2);
    __bf16* ofin = (__bf16*)alloc((size_t)MQ * HDIM * 2);

    // weight convert+transpose to bf16 N-major
    transpose_to_bf16_kernel<<<dim3(HDIM / 32, HDIM / 32), 256, 0, stream>>>(Wq,  WqT,  HDIM, HDIM);
    transpose_to_bf16_kernel<<<dim3(HDIM / 32, HDIM / 32), 256, 0, stream>>>(Wk,  WkT,  HDIM, HDIM);
    transpose_to_bf16_kernel<<<dim3(HDIM / 32, HDIM / 32), 256, 0, stream>>>(Wv,  WvT,  HDIM, HDIM);
    transpose_to_bf16_kernel<<<dim3(HDIM / 32, HDIM / 32), 256, 0, stream>>>(Wo,  WoT,  HDIM, HDIM);
    transpose_to_bf16_kernel<<<dim3(HDIM / 32, VOCAB / 32), 256, 0, stream>>>(Wlm, WlmT, HDIM, VOCAB);

    anchor_kernel<<<BB, 256, 0, stream>>>(input_ids, loss_mask, rand_vals, anch, keep, tokm, ctx);
    build_rows_kernel<<<BB * NA * (BLK + LL + BLK), 256, 0, stream>>>(embed, hidden, tokm, ctx, noiseA, kvA);

    // projections (bf16 WMMA, f32 out)
    gemm_bf16_kernel<0><<<dim3(HDIM / 64, MQ / 64),  128, 0, stream>>>(noiseA, WqT, qf, nullptr, MQ,  HDIM, HDIM, nullptr, nullptr, nullptr);
    gemm_bf16_kernel<0><<<dim3(HDIM / 64, MKV / 64), 128, 0, stream>>>(kvA,    WkT, kf, nullptr, MKV, HDIM, HDIM, nullptr, nullptr, nullptr);
    gemm_bf16_kernel<0><<<dim3(HDIM / 64, MKV / 64), 128, 0, stream>>>(kvA,    WvT, vf, nullptr, MKV, HDIM, HDIM, nullptr, nullptr, nullptr);

    rope_kernel<<<(BB * NA * BLK        * NHH * 64) / 256, 256, 0, stream>>>(qf, BLK,      0, anch, ctx);
    rope_kernel<<<(BB * NA * (LL + BLK) * NHH * 64) / 256, 256, 0, stream>>>(kf, LL + BLK, 1, anch, ctx);

    attn_kernel<<<BB * NA * NHH, 128, 0, stream>>>(qf, kf, vf, keep, oA);

    // Wo projection with fused residual + keep mask -> bf16 rows for LM head
    gemm_bf16_kernel<1><<<dim3(HDIM / 64, MQ / 64), 128, 0, stream>>>(oA, WoT, nullptr, ofin, MQ, HDIM, HDIM, tokm, keep, embed);

    // LM head: (4096 x 2048) * (2048 x 32000) -> fp32 logits
    gemm_bf16_kernel<0><<<dim3(VOCAB / 64, MQ / 64), 128, 0, stream>>>(ofin, WlmT, out, nullptr, MQ, VOCAB, HDIM, nullptr, nullptr, nullptr);
}
